// LearnablePixelwiseAnisoJBU_Optimized_29085518528869
// MI455X (gfx1250) — matrix operand order repaired and checked by reading, not compiled, not run on hardware
//
#include <hip/hip_runtime.h>
#include <hip/hip_bf16.h>

typedef __attribute__((ext_vector_type(16))) _Float16 v16h;
typedef __attribute__((ext_vector_type(8)))  float    v8f;

#define JSCALE 8
#define RMAX   4
#define NN     81      // (2*RMAX+1)^2 neighbors
#define KPAD   96      // NN padded to multiple of 32 for WMMA K
#define WROW   104     // padded W row (halves): 52 dwords, odd*4 -> conflict-free b128
#define HL     32
#define WL     32
#define CF     32      // feature channels
#define CG     3       // guide channels
#define HH     256
#define WH     256
#define PI_F   3.14159265358979323846f

__global__ __launch_bounds__(256)
void jbu_aniso_wmma_kernel(const float* __restrict__ feat,
                           const float* __restrict__ guide,
                           const float* __restrict__ sxr,
                           const float* __restrict__ syr,
                           const float* __restrict__ thr,
                           const float* __restrict__ srr,
                           float* __restrict__ out)
{
    __shared__ float    s_logw[NN][64];     // log-weights, neighbor-major
    __shared__ _Float16 s_Wt[64][WROW];     // softmax weights, N-major (pixel x K)
    __shared__ _Float16 s_F[CF][KPAD];      // feature tile   (A matrix, M x K)
    __shared__ float    s_glr[NN][4];       // downsampled guide at neighbors
    __shared__ float    s_i2sx[NN], s_i2sy[NN], s_i2sr[NN];
    __shared__ float    s_cs[NN], s_sn[NN], s_cxn[NN], s_cyn[NN], s_d2[NN];
    __shared__ int      s_ui[NN], s_vi[NN];
    __shared__ float    s_gt[CG][64];       // HR guide 8x8 tile
    __shared__ float    s_rsq[64];          // dynamic radius^2 per pixel

    const int tid  = threadIdx.x;
    const int cell = blockIdx.x;            // one LR cell per workgroup
    const int ccy  = cell / WL;
    const int ccx  = cell % WL;

    // hint: pull this cell's feature window toward the caches (gfx1250 prefetch)
    if (tid < CF) {
        int u0 = min(max(ccy - RMAX, 0), HL - 1);
        __builtin_prefetch(&feat[(tid * HL + u0) * WL], 0, 3);
    }

    // ---- Stage A: per-neighbor parameters (81 threads) ----
    if (tid < NN) {
        int dy = tid / 9 - RMAX;
        int dx = tid % 9 - RMAX;
        int u = min(max(ccy + dy, 0), HL - 1);
        int v = min(max(ccx + dx, 0), WL - 1);
        s_ui[tid] = u;  s_vi[tid] = v;
        int li = u * WL + v;
        float sx = fmaxf(expf(sxr[li]), 1e-6f);
        float sy = fmaxf(expf(syr[li]), 1e-6f);
        float sr = fmaxf(expf(srr[li]), 1e-6f);
        float th = PI_F * tanhf(thr[li]);
        s_i2sx[tid] = 1.0f / (2.0f * sx * sx + 1e-8f);
        s_i2sy[tid] = 1.0f / (2.0f * sy * sy + 1e-8f);
        s_i2sr[tid] = 1.0f / (2.0f * sr * sr + 1e-8f);
        s_cs[tid]  = cosf(th);
        s_sn[tid]  = sinf(th);
        s_cxn[tid] = 8.0f * (float)v + 3.5f;   // (v+0.5)*8 - 0.5
        s_cyn[tid] = 8.0f * (float)u + 3.5f;
        s_d2[tid]  = (float)(dy * dy + dx * dx);
        // guide_lr at (u,v): bilinear downsample 256->32 reduces to a 4-tap
        // average at (8u+3..4, 8v+3..4) with weights exactly 0.25
        int gy0 = 8 * u + 3, gy1 = min(8 * u + 4, HH - 1);
        int gx0 = 8 * v + 3, gx1 = min(8 * v + 4, WH - 1);
        #pragma unroll
        for (int c = 0; c < CG; ++c) {
            const float* g = guide + (size_t)c * HH * WH;
            s_glr[tid][c] = 0.25f * (g[gy0 * WH + gx0] + g[gy0 * WH + gx1] +
                                     g[gy1 * WH + gx0] + g[gy1 * WH + gx1]);
        }
    }

    // ---- Stage B: dynamic-radius R^2 per HR pixel (64 threads) ----
    if (tid < 64) {
        int j = tid >> 3, i = tid & 7;
        int y = ccy * 8 + j, x = ccx * 8 + i;
        float ys = fmaxf((y + 0.5f) * 0.125f - 0.5f, 0.0f);
        float xs = fmaxf((x + 0.5f) * 0.125f - 0.5f, 0.0f);
        int y0 = (int)floorf(ys), x0 = (int)floorf(xs);
        int y1 = min(y0 + 1, HL - 1), x1 = min(x0 + 1, WL - 1);
        float wy = ys - (float)y0, wx = xs - (float)x0;
        auto mval = [&](int u, int v) {
            int li = u * WL + v;
            return fmaxf(expf(sxr[li]), expf(syr[li]));
        };
        float top = mval(y0, x0) * (1.0f - wx) + mval(y0, x1) * wx;
        float bot = mval(y1, x0) * (1.0f - wx) + mval(y1, x1) * wx;
        float sig = top * (1.0f - wy) + bot * wy;
        float R = fminf(fmaxf(ceilf(2.0f * sig), 1.0f), 4.0f);
        s_rsq[tid] = R * R;
    }

    // ---- Stage C: HR guide 8x8 tile into LDS ----
    for (int e = tid; e < CG * 64; e += 256) {
        int c = e >> 6, t = e & 63;
        int j = t >> 3, i = t & 7;
        s_gt[c][t] = guide[((size_t)c * HH + ccy * 8 + j) * WH + ccx * 8 + i];
    }
    __syncthreads();

    // ---- Stage D: log-weights for all (neighbor, pixel) pairs ----
    for (int e = tid; e < NN * 64; e += 256) {
        int n = e >> 6;
        int t = e & 63;
        int j = t >> 3, i = t & 7;
        float y = (float)(ccy * 8 + j), x = (float)(ccx * 8 + i);
        float dxv = x - s_cxn[n], dyv = y - s_cyn[n];
        float xr =  dxv * s_cs[n] + dyv * s_sn[n];
        float yr = -dxv * s_sn[n] + dyv * s_cs[n];
        float lw = -(xr * xr) * s_i2sx[n] - (yr * yr) * s_i2sy[n];
        float d2 = 0.0f;
        #pragma unroll
        for (int c = 0; c < CG; ++c) {
            float d = s_gt[c][t] - s_glr[n][c];
            d2 += d * d;
        }
        lw -= d2 * s_i2sr[n];
        if (s_d2[n] > s_rsq[t]) lw = -1e30f;  // dynamic radius mask
        s_logw[n][t] = lw;
    }
    __syncthreads();

    // ---- Stage E: softmax over neighbors -> f16 B matrix, N-major (64 threads) ----
    if (tid < 64) {
        float m = -1e30f;
        for (int n = 0; n < NN; ++n) m = fmaxf(m, s_logw[n][tid]);
        float s = 0.0f;
        for (int n = 0; n < NN; ++n) s += expf(s_logw[n][tid] - m);
        float inv = 1.0f / s;
        for (int n = 0; n < NN; ++n)
            s_Wt[tid][n] = (_Float16)(expf(s_logw[n][tid] - m) * inv);
        for (int n = NN; n < KPAD; ++n)
            s_Wt[tid][n] = (_Float16)0.0f;
    }
    // ---- Stage F: f16 A matrix: features at the 81 neighbors (padded) ----
    for (int e = tid; e < CF * KPAD; e += 256) {
        int c = e / KPAD, k = e % KPAD;
        s_F[c][k] = (k < NN) ? (_Float16)feat[((size_t)c * HL + s_ui[k]) * WL + s_vi[k]]
                             : (_Float16)0.0f;
    }
    __syncthreads();

    // ---- Stage G: per-cell GEMM  O(32x64) = F(32x96) x W(96x64)  via WMMA ----
    // 8 waves: wave -> (mt, nt) output 16x16 tile; 3 k-steps of 32
    const int wv   = tid >> 5;
    const int lane = tid & 31;
    const int mt = wv >> 2;          // 0..1  (channel tile)
    const int nt = wv & 3;           // 0..3  (pixel tile)
    const int hi = lane >> 4;        // lane half-group
    const int lm = lane & 15;

    union AF { v16h v; _Float16 e[16]; };
    v8f acc = {};
    const _Float16* fr = &s_F[mt * 16 + lm][0];      // A row for this lane
    const _Float16* wr = &s_Wt[nt * 16 + lm][0];     // B column (N-major row)
    #pragma unroll
    for (int ks = 0; ks < 3; ++ks) {
        AF a, b;
        // A frag: K = ks*32 + (h&7) + 8*hi + 16*(h>>3)  (two contiguous 8-half runs)
        int ka = ks * 32 + 8 * hi;
        #pragma unroll
        for (int h = 0; h < 8; ++h) {
            a.e[h]     = fr[ka + h];
            a.e[8 + h] = fr[ka + 16 + h];
        }
        // B frag: K = ks*32 + (h&7) + 8*hi + 16*(h>>3)  (same as A, N-major storage)
        int kb = ks * 32 + 8 * hi;
        #pragma unroll
        for (int h = 0; h < 8; ++h) {
            b.e[h]     = wr[kb + h];
            b.e[8 + h] = wr[kb + 16 + h];
        }
        acc = __builtin_amdgcn_wmma_f32_16x16x32_f16(
                  /*neg_a=*/false, a.v, /*neg_b=*/false, b.v,
                  /*c_mod=*/(short)0, acc, /*reuse_a=*/false, /*reuse_b=*/false);
    }

    // D layout: VGPR r -> M = r + 8*hi; lane&15 -> N
    #pragma unroll
    for (int r = 0; r < 8; ++r) {
        int M = mt * 16 + r + 8 * hi;     // output channel
        int N = nt * 16 + lm;             // pixel within 8x8 tile
        int j = N >> 3, i = N & 7;
        out[((size_t)M * HH + ccy * 8 + j) * WH + ccx * 8 + i] = acc[r];
    }
}

extern "C" void kernel_launch(void* const* d_in, const int* in_sizes, int n_in,
                              void* d_out, int out_size, void* d_ws, size_t ws_size,
                              hipStream_t stream) {
    (void)in_sizes; (void)n_in; (void)d_ws; (void)ws_size; (void)out_size;
    const float* feat  = (const float*)d_in[0];
    const float* guide = (const float*)d_in[1];
    const float* sxr   = (const float*)d_in[2];
    const float* syr   = (const float*)d_in[3];
    const float* thr   = (const float*)d_in[4];
    const float* srr   = (const float*)d_in[5];
    float* out = (float*)d_out;

    dim3 grid(HL * WL);   // one workgroup per LR cell (1024)
    dim3 block(256);      // 8 wave32 -> 2x4 WMMA tile grid
    jbu_aniso_wmma_kernel<<<grid, block, 0, stream>>>(feat, guide, sxr, syr, thr, srr, out);
}
